// GNNCritic_60258391162971
// MI455X (gfx1250) — matrix operand order repaired and checked by reading, not compiled, not run on hardware
//
#include <hip/hip_runtime.h>
#include <hip/hip_bf16.h>

#define CD 256

typedef __attribute__((ext_vector_type(16))) _Float16 v16h;
typedef __attribute__((ext_vector_type(8)))  _Float16 v8h;
typedef __attribute__((ext_vector_type(8)))  float    v8f;

// ---------------------------------------------------------------------------
// init: deg=1 (self loop), pooled=0, pack W_gcn (f32->f16) into per-lane
// B-fragment order so the GEMM can load each WMMA B operand as one v16h.
// Packed index t = ((nt*8 + k)*32 + lane)*16 + i  maps to
// W[(k*32 + (lane>>4)*8 + (i&7) + (i>>3)*16)*256 + nt*16 + (lane&15)]
// ---------------------------------------------------------------------------
__global__ __launch_bounds__(256) void k_init(float* __restrict__ deg,
                                              float* __restrict__ pooled,
                                              const float* __restrict__ Wg,
                                              _Float16* __restrict__ Wpack, int n) {
  int i = blockIdx.x * 256 + threadIdx.x;
  if (i < n) deg[i] = 1.0f;
  if (i < 32) pooled[i] = 0.0f;
  if (i < CD * CD) {
    int ii = i & 15;
    int lane = (i >> 4) & 31;
    int k = (i >> 9) & 7;
    int nt = i >> 12;
    int g = lane >> 4, q = lane & 15;
    Wpack[i] = (_Float16)Wg[(k * 32 + g * 8 + (ii & 7) + (ii >> 3) * 16) * CD + nt * 16 + q];
  }
}

__global__ __launch_bounds__(256) void k_deg(const long long* __restrict__ dst,
                                             float* __restrict__ deg, int E) {
  int e = blockIdx.x * 256 + threadIdx.x;
  if (e < E) atomicAdd(&deg[dst[e]], 1.0f);
}

__global__ __launch_bounds__(256) void k_dinv(float* __restrict__ deg, int n) {
  int i = blockIdx.x * 256 + threadIdx.x;
  if (i < n) deg[i] = rsqrtf(deg[i]);
}

// ---------------------------------------------------------------------------
// xw = x @ W_gcn  via v_wmma_f32_16x16x32_f16, with the self-loop epilogue
// fused in:  agg = xw * dinv[row]^2  is stored alongside xw (removes an
// entire 100MB read+write pass). B fragments are manually double-buffered
// (bA/bB both live) so loads for tile nt+1 are in flight while the WMMA
// chain for tile nt executes -- the allocator cannot re-fuse load+wmma.
// ---------------------------------------------------------------------------
__global__ __launch_bounds__(256) void k_gemm(const float* __restrict__ x,
                                              const _Float16* __restrict__ Wpack,
                                              const float* __restrict__ dinv,
                                              float* __restrict__ xw,
                                              float* __restrict__ agg, int n) {
  const int lane = threadIdx.x & 31;
  const int wave = threadIdx.x >> 5;
  const int g = lane >> 4, q = lane & 15;
  const long long rowBase = (long long)blockIdx.x * 128 + wave * 16;
  const bool full = (rowBase + 16 <= (long long)n);   // wave-uniform

  long long r = rowBase + q;
  if (r > n - 1) r = n - 1;                    // clamp: keeps EXEC uniform
  const float4* xr = (const float4*)(x + r * CD);

  // dinv^2 for this lane's 8 output rows (rowBase + g*8 + rr)
  float dsc[8];
#pragma unroll
  for (int rr = 0; rr < 8; ++rr) {
    long long rg = rowBase + g * 8 + rr;
    if (rg > n - 1) rg = n - 1;
    float di = dinv[rg];
    dsc[rr] = di * di;
  }

  v16h a[8];                                   // A frags for all 8 K-steps
#pragma unroll
  for (int k = 0; k < 8; ++k) {
    float4 p0 = xr[k * 8 + g * 2];
    float4 p1 = xr[k * 8 + g * 2 + 1];
    float4 p2 = xr[k * 8 + 4 + g * 2];
    float4 p3 = xr[k * 8 + 4 + g * 2 + 1];
    a[k][0]  = (_Float16)p0.x; a[k][1]  = (_Float16)p0.y;
    a[k][2]  = (_Float16)p0.z; a[k][3]  = (_Float16)p0.w;
    a[k][4]  = (_Float16)p1.x; a[k][5]  = (_Float16)p1.y;
    a[k][6]  = (_Float16)p1.z; a[k][7]  = (_Float16)p1.w;
    a[k][8]  = (_Float16)p2.x; a[k][9]  = (_Float16)p2.y;
    a[k][10] = (_Float16)p2.z; a[k][11] = (_Float16)p2.w;
    a[k][12] = (_Float16)p3.x; a[k][13] = (_Float16)p3.y;
    a[k][14] = (_Float16)p3.z; a[k][15] = (_Float16)p3.w;
  }

#define STORE_TILE(NT, ACC)                                               \
  {                                                                       \
    float* o  = xw  + (rowBase + g * 8) * CD + (NT) * 16 + q;             \
    float* oa = agg + (rowBase + g * 8) * CD + (NT) * 16 + q;             \
    if (full) {                                                           \
      _Pragma("unroll")                                                   \
      for (int rr = 0; rr < 8; ++rr) {                                    \
        o[rr * CD]  = ACC[rr];                                            \
        oa[rr * CD] = ACC[rr] * dsc[rr];                                  \
      }                                                                   \
    } else {                                                              \
      _Pragma("unroll")                                                   \
      for (int rr = 0; rr < 8; ++rr)                                      \
        if (rowBase + g * 8 + rr < n) {                                   \
          o[rr * CD]  = ACC[rr];                                          \
          oa[rr * CD] = ACC[rr] * dsc[rr];                                \
        }                                                                 \
    }                                                                     \
  }

  v16h bA[8], bB[8];
#pragma unroll
  for (int k = 0; k < 8; ++k)
    bA[k] = *(const v16h*)(Wpack + ((k * 32 + lane) << 4));

#pragma unroll 1
  for (int nt = 0; nt < 16; nt += 2) {
    // issue loads for tile nt+1 while computing tile nt from bA
#pragma unroll
    for (int k = 0; k < 8; ++k)
      bB[k] = *(const v16h*)(Wpack + ((((nt + 1) * 8 + k) * 32 + lane) << 4));

    v8f acc = {};
#pragma unroll
    for (int k = 0; k < 8; ++k)
      acc = __builtin_amdgcn_wmma_f32_16x16x32_f16(false, a[k], false, bA[k],
                                                   (short)0, acc, false, false);
    STORE_TILE(nt, acc);

    // issue loads for tile nt+2 while computing tile nt+1 from bB
    if (nt + 2 < 16) {
#pragma unroll
      for (int k = 0; k < 8; ++k)
        bA[k] = *(const v16h*)(Wpack + ((((nt + 2) * 8 + k) * 32 + lane) << 4));
    }

    v8f acc2 = {};
#pragma unroll
    for (int k = 0; k < 8; ++k)
      acc2 = __builtin_amdgcn_wmma_f32_16x16x32_f16(false, a[k], false, bB[k],
                                                    (short)0, acc2, false, false);
    STORE_TILE(nt + 1, acc2);
  }
#undef STORE_TILE
}

// one block per edge: coalesced 256-wide gather-scale-scatter (L2 atomics)
__global__ __launch_bounds__(256) void k_edge(const long long* __restrict__ src,
                                              const long long* __restrict__ dst,
                                              const float* __restrict__ xw,
                                              const float* __restrict__ dinv,
                                              float* __restrict__ agg) {
  int e = blockIdx.x;
  long long s = src[e], d = dst[e];
  float nrm = dinv[s] * dinv[d];
  int c = threadIdx.x;
  atomicAdd(&agg[d * CD + c], xw[s * CD + c] * nrm);
}

// ---------------------------------------------------------------------------
// Fused node head:  h = relu(agg + b_gcn) + x   (built on the fly into f16
// A-fragments), then [h|action|price] @ W1 (+relu) @ W2 (+relu), sum-pool.
// Both small GEMMs use WMMA; W1/W2 are packed into B-fragment order in LDS.
// ---------------------------------------------------------------------------
__global__ __launch_bounds__(256) void k_mlp(const float* __restrict__ x,
                                             const float* __restrict__ agg,
                                             const float* __restrict__ bg,
                                             const float* __restrict__ action,
                                             const float* __restrict__ price,
                                             const float* __restrict__ W1,
                                             const float* __restrict__ b1,
                                             const float* __restrict__ W2,
                                             const float* __restrict__ b2,
                                             float* __restrict__ pooled, int n) {
  __shared__ __align__(32) _Float16 sW1[2 * 8 * 32 * 16];  // 16 KB packed B frags
  __shared__ __align__(32) _Float16 sW2[2 * 32 * 16];      // 2 KB packed B frags
  __shared__ __align__(16) float sbg[CD];
  __shared__ float sW1a[32], sW1p[32], sb1[32], sb2[32];
  __shared__ __align__(32) _Float16 sz[8][16 * 32];        // per-wave z staging

  for (int t = threadIdx.x; t < 8192; t += 256) {
    int ii = t & 15, ln = (t >> 4) & 31, k = (t >> 9) & 7, nt = t >> 12;
    sW1[t] = (_Float16)W1[(k * 32 + (ln >> 4) * 8 + (ii & 7) + (ii >> 3) * 16) * 32 + nt * 16 + (ln & 15)];
  }
  for (int t = threadIdx.x; t < 1024; t += 256) {
    int ii = t & 15, ln = (t >> 4) & 31, nt = t >> 9;
    sW2[t] = (_Float16)W2[((ln >> 4) * 8 + (ii & 7) + (ii >> 3) * 16) * 32 + nt * 16 + (ln & 15)];
  }
  sbg[threadIdx.x] = bg[threadIdx.x];
  if (threadIdx.x < 32) {
    sW1a[threadIdx.x] = W1[256 * 32 + threadIdx.x];   // action row of W1
    sW1p[threadIdx.x] = W1[257 * 32 + threadIdx.x];   // price  row of W1
    sb1[threadIdx.x] = b1[threadIdx.x];
    sb2[threadIdx.x] = b2[threadIdx.x];
  }
  __syncthreads();

  const int lane = threadIdx.x & 31;
  const int wave = threadIdx.x >> 5;
  const int g = lane >> 4, q = lane & 15;
  const long long rowBase = (long long)blockIdx.x * 128 + wave * 16;
  const bool full = (rowBase + 16 <= (long long)n);   // wave-uniform

  long long r = rowBase + q;
  if (r > n - 1) r = n - 1;
  const float4* agr = (const float4*)(agg + r * CD);
  const float4* xr  = (const float4*)(x + r * CD);
  const float4* sb4 = (const float4*)sbg;

#define HQUAD(K, D, AV, XV, CV)                                   \
  a[K][(D) + 0] = (_Float16)(fmaxf(AV.x + CV.x, 0.f) + XV.x);     \
  a[K][(D) + 1] = (_Float16)(fmaxf(AV.y + CV.y, 0.f) + XV.y);     \
  a[K][(D) + 2] = (_Float16)(fmaxf(AV.z + CV.z, 0.f) + XV.z);     \
  a[K][(D) + 3] = (_Float16)(fmaxf(AV.w + CV.w, 0.f) + XV.w)

  v16h a[8];
#pragma unroll
  for (int k = 0; k < 8; ++k) {
    int i0 = k * 8 + g * 2;
    int i1 = k * 8 + 4 + g * 2;
    float4 g0 = agr[i0], g1 = agr[i0 + 1], g2 = agr[i1], g3 = agr[i1 + 1];
    float4 x0 = xr[i0],  x1 = xr[i0 + 1],  x2 = xr[i1],  x3 = xr[i1 + 1];
    float4 c0 = sb4[i0], c1 = sb4[i0 + 1], c2 = sb4[i1], c3 = sb4[i1 + 1];
    HQUAD(k, 0,  g0, x0, c0);
    HQUAD(k, 4,  g1, x1, c1);
    HQUAD(k, 8,  g2, x2, c2);
    HQUAD(k, 12, g3, x3, c3);
  }
#undef HQUAD

  float pr = price[0];
  _Float16* zs = sz[wave];
#pragma unroll
  for (int nt = 0; nt < 2; ++nt) {
    v16h b[8];
#pragma unroll
    for (int k = 0; k < 8; ++k)
      b[k] = *(const v16h*)(sW1 + (((nt * 8 + k) * 32 + lane) << 4));

    v8f acc = {};
#pragma unroll
    for (int k = 0; k < 8; ++k)
      acc = __builtin_amdgcn_wmma_f32_16x16x32_f16(false, a[k], false, b[k],
                                                   (short)0, acc, false, false);

    int j = nt * 16 + q;
    float wa = sW1a[j], wp = sW1p[j], bb = sb1[j];
    if (full) {
#pragma unroll
      for (int rr = 0; rr < 8; ++rr) {
        float z = acc[rr] + bb + action[rowBase + g * 8 + rr] * wa + pr * wp;
        zs[(g * 8 + rr) * 32 + j] = (_Float16)fmaxf(z, 0.f);
      }
    } else {
#pragma unroll
      for (int rr = 0; rr < 8; ++rr) {
        long long rg = rowBase + g * 8 + rr;
        if (rg > n - 1) rg = n - 1;
        float z = acc[rr] + bb + action[rg] * wa + pr * wp;
        zs[(g * 8 + rr) * 32 + j] = (_Float16)fmaxf(z, 0.f);
      }
    }
  }
  __syncthreads();

  v16h a2;
  {
    v8h lo = *(const v8h*)(zs + q * 32 + g * 8);
    v8h hi = *(const v8h*)(zs + q * 32 + 16 + g * 8);
#pragma unroll
    for (int i = 0; i < 8; ++i) { a2[i] = lo[i]; a2[8 + i] = hi[i]; }
  }

  float csum0 = 0.f, csum1 = 0.f;
#pragma unroll
  for (int nt = 0; nt < 2; ++nt) {
    v16h b = *(const v16h*)(sW2 + ((nt * 32 + lane) << 4));
    v8f acc = {};
    acc = __builtin_amdgcn_wmma_f32_16x16x32_f16(false, a2, false, b,
                                                 (short)0, acc, false, false);
    int j = nt * 16 + q;
    float bb = sb2[j];
    float s = 0.f;
    if (full) {
#pragma unroll
      for (int rr = 0; rr < 8; ++rr) s += fmaxf(acc[rr] + bb, 0.f);
    } else {
#pragma unroll
      for (int rr = 0; rr < 8; ++rr) {
        float z2 = fmaxf(acc[rr] + bb, 0.f);
        if (rowBase + g * 8 + rr < n) s += z2;  // exclude clamped ghost rows
      }
    }
    if (nt == 0) csum0 = s; else csum1 = s;
  }
  csum0 += __shfl_xor(csum0, 16, 32);   // merge M=0..7 / M=8..15 half-waves
  csum1 += __shfl_xor(csum1, 16, 32);
  if (lane < 16) {
    atomicAdd(&pooled[lane],      csum0);
    atomicAdd(&pooled[16 + lane], csum1);
  }
}

__global__ void k_final(const float* __restrict__ pooled,
                        const float* __restrict__ W3,
                        const float* __restrict__ b3,
                        float* __restrict__ out) {
  if (threadIdx.x == 0) {
    float s = b3[0];
    for (int j = 0; j < 32; ++j) s += pooled[j] * W3[j];
    out[0] = s;
  }
}

extern "C" void kernel_launch(void* const* d_in, const int* in_sizes, int n_in,
                              void* d_out, int out_size, void* d_ws, size_t ws_size,
                              hipStream_t stream) {
  const float* x        = (const float*)d_in[0];
  const long long* ei   = (const long long*)d_in[1];
  const float* action   = (const float*)d_in[2];
  const float* price    = (const float*)d_in[3];
  const float* Wg       = (const float*)d_in[4];
  const float* bg       = (const float*)d_in[5];
  const float* W1       = (const float*)d_in[6];
  const float* b1       = (const float*)d_in[7];
  const float* W2       = (const float*)d_in[8];
  const float* b2       = (const float*)d_in[9];
  const float* W3       = (const float*)d_in[10];
  const float* b3       = (const float*)d_in[11];
  const int n = in_sizes[0] / CD;
  const int E = in_sizes[1] / 2;
  const long long* src = ei;
  const long long* dst = ei + E;

  // workspace carve-up (float units, 256B-aligned chunks)
  float* ws = (float*)d_ws;
  size_t off = 0;
  float* deg = ws;                         off += ((size_t)n + 63) & ~(size_t)63;
  float* pooled = ws + off;                off += 64;
  _Float16* Wpack = (_Float16*)(ws + off); off += (size_t)(CD * CD) / 2;
  float* xw = ws + off;                    off += (size_t)n * CD;
  float* agg = ws + off;

  int initN = n > CD * CD ? n : CD * CD;
  k_init<<<(initN + 255) / 256, 256, 0, stream>>>(deg, pooled, Wg, Wpack, n);
  k_deg<<<(E + 255) / 256, 256, 0, stream>>>(dst, deg, E);
  k_dinv<<<(n + 255) / 256, 256, 0, stream>>>(deg, n);
  k_gemm<<<(n + 127) / 128, 256, 0, stream>>>(x, Wpack, deg, xw, agg, n);
  k_edge<<<E, 256, 0, stream>>>(src, dst, xw, deg, agg);
  k_mlp<<<(n + 127) / 128, 256, 0, stream>>>(x, agg, bg, action, price,
                                             W1, b1, W2, b2, pooled, n);
  k_final<<<1, 64, 0, stream>>>(pooled, W3, b3, (float*)d_out);
}
